// CRFDecoder_41721312314027
// MI455X (gfx1250) — matrix operand order repaired and checked by reading, not compiled, not run on hardware
//
#include <hip/hip_runtime.h>
#include <hip/hip_bf16.h>

#define BB 32
#define TT 512
#define DD 512
#define VV 512

typedef __attribute__((ext_vector_type(16))) __bf16 v16bf;
typedef __attribute__((ext_vector_type(8)))  __bf16 v8bf;
typedef __attribute__((ext_vector_type(8)))  float  v8f;

// ---------------- TDM availability / arity hedge ----------------
#if defined(__has_builtin)
#if __has_builtin(__builtin_amdgcn_tensor_load_to_lds) && __has_builtin(__builtin_amdgcn_s_wait_tensorcnt)
#define HAVE_TDM 1
#endif
#endif
#ifndef HAVE_TDM
#define HAVE_TDM 0
#endif
#if __has_include(<hip/amd_detail/amd_gfx1250_TDM.h>)
#define TDM_SIX_ARGS 1
#else
#define TDM_SIX_ARGS 0
#endif

#if HAVE_TDM
typedef __attribute__((ext_vector_type(4))) unsigned u32x4;
typedef __attribute__((ext_vector_type(8))) int      i32x8;
typedef __attribute__((ext_vector_type(4))) int      i32x4;

// Issue a TDM load of one contiguous row of 512 f32 (2KB) into LDS.
// D# per CDNA5 ISA §8: group0 = {count=1 | lds_addr | global_addr | type=2},
// group1 = {data_size=4B, tensor_dim0=512, tensor_dim1=1, tile=512x1, stride=512}.
__device__ inline void tdm_row_load(const float* src, unsigned lds_off) {
    unsigned long long ga = (unsigned long long)src;
    u32x4 g0;
    g0.x = 1u;                                                   // count=1 (valid, user)
    g0.y = lds_off;                                              // lds_addr (bytes)
    g0.z = (unsigned)ga;                                         // global_addr[31:0]
    g0.w = (unsigned)((ga >> 32) & 0x01FFFFFFu) | 0x80000000u;   // addr[56:32] | type=2
    i32x8 g1;
    g1.s0 = 0x00020000;            // workgroup_mask=0, data_size=2 (4 bytes)
    g1.s1 = (int)(512u << 16);     // tensor_dim0 = 512
    g1.s2 = (int)(1u << 16);       // tensor_dim1 = 1
    g1.s3 = (int)(512u << 16);     // tile_dim0 = 512
    g1.s4 = 1;                     // tile_dim1 = 1 (tile_dim2 = 0)
    g1.s5 = 512;                   // tensor_dim0_stride = 512
    g1.s6 = 0;
    g1.s7 = 0;
    i32x4 z4 = {0, 0, 0, 0};
#if TDM_SIX_ARGS
    i32x8 z8 = {0, 0, 0, 0, 0, 0, 0, 0};
    __builtin_amdgcn_tensor_load_to_lds(g0, g1, z4, z4, z8, 0);
#else
    __builtin_amdgcn_tensor_load_to_lds(g0, g1, z4, z4, 0);
#endif
}
#endif // HAVE_TDM

// =====================================================================
// Prep kernels: one-shot bf16 hi/lo splits (removes per-tile VALU split
// work from the GEMM inner loop entirely).
// =====================================================================
__global__ void split_x(const float* __restrict__ X,
                        __bf16* __restrict__ Xh, __bf16* __restrict__ Xl) {
    int i = blockIdx.x * 256 + threadIdx.x;   // BB*TT*DD elements
    float f = X[i];
    unsigned u = __builtin_bit_cast(unsigned, f);
    float hif = __builtin_bit_cast(float, u & 0xFFFF0000u);
    float rem = f - hif;
    Xh[i] = __builtin_bit_cast(__bf16, (unsigned short)(u >> 16));
    unsigned ur = __builtin_bit_cast(unsigned, rem);
    Xl[i] = __builtin_bit_cast(__bf16, (unsigned short)(ur >> 16));
}

// W (V,D) -> bf16 splits WhT/WlT, same (V,D) layout: B-fragments become
// contiguous 16B loads in the GEMM.
__global__ void split_wT(const float* __restrict__ W,
                         __bf16* __restrict__ WhT, __bf16* __restrict__ WlT) {
    int i = blockIdx.x * 256 + threadIdx.x;   // VV*DD elements
    int k = i >> 9;          // 0..511 (V index)
    int v = i & 511;         // 0..511 (D index)
    float f = W[(size_t)k * DD + v];
    unsigned u = __builtin_bit_cast(unsigned, f);
    float hif = __builtin_bit_cast(float, u & 0xFFFF0000u);
    float rem = f - hif;
    WhT[i] = __builtin_bit_cast(__bf16, (unsigned short)(u >> 16));
    unsigned ur = __builtin_bit_cast(unsigned, rem);
    WlT[i] = __builtin_bit_cast(__bf16, (unsigned short)(ur >> 16));
}

__global__ void prep_exp(const float* __restrict__ t, float* __restrict__ e) {
    int i = blockIdx.x * 256 + threadIdx.x;
    e[i] = __expf(t[i]);
}

#define CAT16(a, b) __builtin_shufflevector(a, b, 0,1,2,3,4,5,6,7,8,9,10,11,12,13,14,15)

// =====================================================================
// Kernel: emissions GEMM via bf16x3 split WMMA, pre-split operands.
//   grid (M/64, V/64), block 128 (4 waves); wave -> 16x64 output strip
//   (4 N-tiles reuse one A fragment) -> 12 WMMAs per 32-wide k-chunk.
// =====================================================================
__global__ __launch_bounds__(128) void emis_gemm(
    const __bf16* __restrict__ Xh, const __bf16* __restrict__ Xl,
    const __bf16* __restrict__ WhT, const __bf16* __restrict__ WlT,
    float* __restrict__ E)
{
    const int lane = threadIdx.x & 31;
    const int wave = threadIdx.x >> 5;
    const int l = lane & 15;
    const int h = lane >> 4;
    const int m0 = (blockIdx.x * 4 + wave) * 16;
    const int v0 = blockIdx.y * 64;

    v8f c[4];
#pragma unroll
    for (int j = 0; j < 4; ++j) c[j] = (v8f){0.f,0.f,0.f,0.f,0.f,0.f,0.f,0.f};

    const v8bf* arow_h = (const v8bf*)(Xh + (size_t)(m0 + l) * DD);
    const v8bf* arow_l = (const v8bf*)(Xl + (size_t)(m0 + l) * DD);

    for (int k0 = 0; k0 < DD; k0 += 32) {
        // A fragment: lane (l,h) = row m0+l, k in [k0+8h, +8) U [k0+16+8h, +8)
        const int base = (k0 + 8 * h) >> 3;            // in v8bf units
        v16bf ah = CAT16(arow_h[base], arow_h[base + 2]);
        v16bf al = CAT16(arow_l[base], arow_l[base + 2]);
        // B fragments: lane = N col v0+16j+l, k in [k0+8h, +8) U [k0+16+8h, +8)
        const size_t kb = (size_t)(v0 + l) * DD;
        const v8bf* bhp = (const v8bf*)(WhT + kb) + base;
        const v8bf* blp = (const v8bf*)(WlT + kb) + base;
#pragma unroll
        for (int j = 0; j < 4; ++j) {
            v16bf bh = CAT16(bhp[j * 2 * DD], bhp[j * 2 * DD + 2]);
            v16bf bl = CAT16(blp[j * 2 * DD], blp[j * 2 * DD + 2]);
            c[j] = __builtin_amdgcn_wmma_f32_16x16x32_bf16(false, ah, false, bh, (short)0, c[j], false, false);
            c[j] = __builtin_amdgcn_wmma_f32_16x16x32_bf16(false, al, false, bh, (short)0, c[j], false, false);
            c[j] = __builtin_amdgcn_wmma_f32_16x16x32_bf16(false, ah, false, bl, (short)0, c[j], false, false);
        }
    }
    // C layout: lane (l,h), vgpr r -> C[M = r + 8h][N = l]
#pragma unroll
    for (int j = 0; j < 4; ++j)
#pragma unroll
        for (int r = 0; r < 8; ++r)
            E[(size_t)(m0 + r + 8 * h) * VV + v0 + j * 16 + l] = c[j][r];
}

// ---------- block reductions (512 threads = 16 waves, wave32) ----------
__device__ inline float blockReduceMax(float v, float* red) {
#pragma unroll
    for (int off = 16; off > 0; off >>= 1) v = fmaxf(v, __shfl_xor(v, off, 32));
    int wid = threadIdx.x >> 5, lid = threadIdx.x & 31;
    __syncthreads();
    if (lid == 0) red[wid] = v;
    __syncthreads();
    if (wid == 0) {
        float r = red[lid & 15];
#pragma unroll
        for (int off = 8; off > 0; off >>= 1) r = fmaxf(r, __shfl_xor(r, off, 32));
        if (lid == 0) red[0] = r;
    }
    __syncthreads();
    float out = red[0];
    __syncthreads();
    return out;
}
__device__ inline float blockReduceSum(float v, float* red) {
#pragma unroll
    for (int off = 16; off > 0; off >>= 1) v += __shfl_xor(v, off, 32);
    int wid = threadIdx.x >> 5, lid = threadIdx.x & 31;
    __syncthreads();
    if (lid == 0) red[wid] = v;
    __syncthreads();
    if (wid == 0) {
        float r = red[lid & 15];
        if (lid >= 16) r = 0.f;
#pragma unroll
        for (int off = 16; off > 0; off >>= 1) r += __shfl_xor(r, off, 32);
        if (lid == 0) red[0] = r;
    }
    __syncthreads();
    float out = red[0];
    __syncthreads();
    return out;
}
__device__ inline int blockArgMax(float v, int idx, float* redf, int* redi) {
#pragma unroll
    for (int off = 16; off > 0; off >>= 1) {
        float ov = __shfl_xor(v, off, 32);
        int   oi = __shfl_xor(idx, off, 32);
        if (ov > v || (ov == v && oi < idx)) { v = ov; idx = oi; }
    }
    int wid = threadIdx.x >> 5, lid = threadIdx.x & 31;
    __syncthreads();
    if (lid == 0) { redf[wid] = v; redi[wid] = idx; }
    __syncthreads();
    if (threadIdx.x == 0) {
        for (int w = 1; w < 16; ++w)
            if (redf[w] > v || (redf[w] == v && redi[w] < idx)) { v = redf[w]; idx = redi[w]; }
        redi[0] = idx;
    }
    __syncthreads();
    return redi[0];
}

// =====================================================================
// Fused scans: blocks 0..31 forward (+score+loglik), 32..63 Viterbi
// (+backtrace). Next emissions row prefetched into LDS via TDM, pipelined
// with s_wait_tensorcnt<=1 so the t+1 DMA overlaps the inner matvec.
// =====================================================================
__global__ __launch_bounds__(512) void crf_scan(
    const float* __restrict__ E,            // emissions (B,T,V)
    const float* __restrict__ trans,        // (V,V)
    const float* __restrict__ Etr,          // exp(trans) (V,V)
    const int*   __restrict__ tgt,          // (B,T)
    const unsigned char* __restrict__ pad,  // (B,T) bool
    float* __restrict__ loglik,             // (B,)
    float* __restrict__ tokensF,            // (B,T) as float
    unsigned short* __restrict__ bp)        // (T-1,B,V)
{
    __shared__ float sh[VV];
    __shared__ float ebuf[2][VV];
    __shared__ float redf[16];
    __shared__ int   redi[16];
    const int tid  = threadIdx.x;           // == state v
    const int wid  = tid >> 5;
    const int role = blockIdx.x >> 5;
    const int b    = blockIdx.x & 31;
    (void)wid; (void)ebuf;

    if (role == 0) {
        // ---------------- gold-path score ----------------
        int   tk = tgt[b * TT + tid];
        float mt = pad[b * TT + tid] ? 0.f : 1.f;
        float sc = E[((size_t)b * TT + tid) * VV + tk] * mt;
        if (tid >= 1) {
            int   tkp = tgt[b * TT + tid - 1];
            float mp  = pad[b * TT + tid - 1] ? 0.f : 1.f;
            sc += trans[(size_t)tkp * VV + tk] * mt * mp;
        }
        float score = blockReduceSum(sc, redf);

        // ---------------- forward scan (linear space) ----------------
        float a = E[((size_t)b * TT) * VV + tid];   // alpha0[v]
#if HAVE_TDM
        if (wid == 0)
            tdm_row_load(&E[((size_t)b * TT + 1) * VV], (unsigned)(size_t)&ebuf[1][0]);
#endif
        for (int t = 1; t < TT; ++t) {
            float mx = blockReduceMax(a, redf);
            sh[tid] = __expf(a - mx);
            __syncthreads();
#if HAVE_TDM
            if (wid == 0 && t + 1 < TT)
                tdm_row_load(&E[((size_t)b * TT + t + 1) * VV],
                             (unsigned)(size_t)&ebuf[(t + 1) & 1][0]);
#endif
            float s = 0.f;
            const float* col = Etr + tid;
#pragma unroll 8
            for (int u = 0; u < VV; ++u) s = __fmaf_rn(sh[u], col[(size_t)u * VV], s);
#if HAVE_TDM
            if (wid == 0) {
                if (t + 1 < TT) __builtin_amdgcn_s_wait_tensorcnt(1);
                else            __builtin_amdgcn_s_wait_tensorcnt(0);
            }
            __syncthreads();
            float e = ebuf[t & 1][tid];
#else
            float e = E[((size_t)b * TT + t) * VV + tid];
#endif
            bool pd  = pad[b * TT + t];
            float an = mx + __logf(s) + e;
            a = pd ? a : an;
            __syncthreads();
        }
        float mx = blockReduceMax(a, redf);
        float se = blockReduceSum(__expf(a - mx), redf);
        if (tid == 0) loglik[b] = score - (mx + __logf(se));
    } else {
        // ---------------- Viterbi scan ----------------
        float a = E[((size_t)b * TT) * VV + tid];
#if HAVE_TDM
        if (wid == 0)
            tdm_row_load(&E[((size_t)b * TT + 1) * VV], (unsigned)(size_t)&ebuf[1][0]);
#endif
        for (int t = 1; t < TT; ++t) {
            sh[tid] = a;
            __syncthreads();
#if HAVE_TDM
            if (wid == 0 && t + 1 < TT)
                tdm_row_load(&E[((size_t)b * TT + t + 1) * VV],
                             (unsigned)(size_t)&ebuf[(t + 1) & 1][0]);
#endif
            float best = -3.402823466e38f;
            int   bu   = 0;
            const float* col = trans + tid;
#pragma unroll 4
            for (int u = 0; u < VV; ++u) {
                float scv = sh[u] + col[(size_t)u * VV];
                if (scv > best) { best = scv; bu = u; }   // strict > keeps first argmax
            }
#if HAVE_TDM
            if (wid == 0) {
                if (t + 1 < TT) __builtin_amdgcn_s_wait_tensorcnt(1);
                else            __builtin_amdgcn_s_wait_tensorcnt(0);
            }
            __syncthreads();
            float e = ebuf[t & 1][tid];
#else
            float e = E[((size_t)b * TT + t) * VV + tid];
#endif
            bool pd = pad[b * TT + t];
            bp[((size_t)(t - 1) * BB + b) * VV + tid] = (unsigned short)(pd ? tid : bu);
            a = pd ? a : (best + e);
            __syncthreads();
        }
        int last = blockArgMax(a, tid, redf, redi);
        if (tid == 0) {
            int cur = last;
            for (int t = TT - 2; t >= 0; --t) {
                tokensF[b * TT + t + 1] = (float)cur;
                cur = bp[((size_t)t * BB + b) * VV + cur];
            }
            tokensF[b * TT + 0] = (float)cur;
        }
    }
}

extern "C" void kernel_launch(void* const* d_in, const int* in_sizes, int n_in,
                              void* d_out, int out_size, void* d_ws, size_t ws_size,
                              hipStream_t stream) {
    (void)in_sizes; (void)n_in; (void)out_size; (void)ws_size;
    const float*         x   = (const float*)d_in[0];          // (B,T,D)
    const float*         w   = (const float*)d_in[1];          // (V,D)
    const float*         tr  = (const float*)d_in[2];          // (V,V)
    const int*           tgt = (const int*)d_in[3];            // (B,T)
    const unsigned char* pad = (const unsigned char*)d_in[4];  // (B,T) bool

    float* emis    = (float*)d_out;                            // (B,T,V)
    float* loglik  = emis + (size_t)BB * TT * VV;              // (B,)
    float* tokensF = loglik + BB;                              // (B,T)

    // workspace layout (bytes)
    char* ws = (char*)d_ws;
    float*          Etr = (float*)(ws);                               // 1,048,576
    unsigned short* bp  = (unsigned short*)(ws + 1048576);            // 16,744,448
    __bf16*         Xh  = (__bf16*)(ws + 17793024);                   // 16,777,216
    __bf16*         Xl  = (__bf16*)(ws + 34570240);                   // 16,777,216
    __bf16*         WhT = (__bf16*)(ws + 51347456);                   //    524,288
    __bf16*         WlT = (__bf16*)(ws + 51871744);                   //    524,288

    const int NX = BB * TT * DD;   // 8,388,608
    const int NW = VV * DD;        //   262,144

    hipLaunchKernelGGL(split_x,  dim3(NX / 256), dim3(256), 0, stream, x, Xh, Xl);
    hipLaunchKernelGGL(split_wT, dim3(NW / 256), dim3(256), 0, stream, w, WhT, WlT);
    hipLaunchKernelGGL(prep_exp, dim3((VV * VV) / 256), dim3(256), 0, stream, tr, Etr);
    hipLaunchKernelGGL(emis_gemm, dim3((BB * TT) / 64, VV / 64), dim3(128), 0, stream,
                       Xh, Xl, WhT, WlT, emis);
    hipLaunchKernelGGL(crf_scan, dim3(64), dim3(512), 0, stream,
                       emis, tr, Etr, tgt, pad, loglik, tokensF, bp);
}
